// diffnetplus_9096740733363
// MI455X (gfx1250) — matrix operand order repaired and checked by reading, not compiled, not run on hardware
//
#include <hip/hip_runtime.h>
#include <hip/hip_bf16.h>
#include <math.h>

typedef __attribute__((ext_vector_type(16))) _Float16 v16h;
typedef __attribute__((ext_vector_type(8)))  float    v8f;

#define FD 64  // feature dim D

// ---------------- utility ----------------
__global__ void fill_f32(float* __restrict__ p, int n, float v) {
    int i = blockIdx.x * blockDim.x + threadIdx.x;
    if (i < n) p[i] = v;
}

// ---------------- sparse softmax (3 passes) ----------------
// pass1: v = exp(sigmoid(raw*w+b)); segment max via uint atomicMax (all values > 0)
__global__ void edge_pass1(const float* __restrict__ raw, const int* __restrict__ row,
                           const float* __restrict__ w_arr, const float* __restrict__ b_arr,
                           int widx, float* __restrict__ v, unsigned* __restrict__ mbits, int E) {
    int e = blockIdx.x * blockDim.x + threadIdx.x;
    if (e >= E) return;
    float w = w_arr[widx], b = b_arr[widx];
    float x  = raw[e] * w + b;
    float sg = 1.0f / (1.0f + __expf(-x));
    float val = __expf(sg);               // in (1, e): positive -> uint-ordered
    v[e] = val;
    atomicMax(&mbits[row[e]], __float_as_uint(val));
}

// pass2: e = exp(v - m[row]); segment sum
__global__ void edge_pass2(float* __restrict__ v, const int* __restrict__ row,
                           const unsigned* __restrict__ mbits, float* __restrict__ s, int E) {
    int e = blockIdx.x * blockDim.x + threadIdx.x;
    if (e >= E) return;
    int r = row[e];
    float m  = __uint_as_float(mbits[r]);
    float ex = __expf(v[e] - m);
    v[e] = ex;
    atomicAdd(&s[r], ex);
}

// pass3: att = e / (s[row] + 1e-12)
__global__ void edge_pass3(float* __restrict__ v, const int* __restrict__ row,
                           const float* __restrict__ s, int E) {
    int e = blockIdx.x * blockDim.x + threadIdx.x;
    if (e >= E) return;
    v[e] = v[e] / (s[row[e]] + 1e-12f);
}

// ---------------- SpMM scatter: out[row[e],:] += att[e] * x[col[e],:] ----------------
// one thread per (edge, 4-feature quad): b128 gather of x, 4 f32 atomics
__global__ void spmm_scatter(const float* __restrict__ att, const int* __restrict__ row,
                             const int* __restrict__ col, const float* __restrict__ x,
                             float* __restrict__ out, int E) {
    int t = blockIdx.x * blockDim.x + threadIdx.x;
    if (t >= E * (FD / 4)) return;
    int e = t >> 4;                 // FD/4 == 16 quads per edge
    int q = (t & 15) << 2;
    float a = att[e];
    const float4 xv = *(const float4*)&x[col[e] * FD + q];
    float* o = &out[row[e] * FD + q];
    atomicAdd(o + 0, a * xv.x);
    atomicAdd(o + 1, a * xv.y);
    atomicAdd(o + 2, a * xv.z);
    atomicAdd(o + 3, a * xv.w);
}

// ---------------- fused graph attention + gated combine (WMMA) ----------------
// Virtual A tile [16 x 192] = [M0 | M1 | M2]; B [192 x 2] cols:
//   col0 = [W_k0[0:64],   W_k0[64:128], 0          ]   (X = concat(M0, M1))
//   col1 = [W_k0+1[0:64], 0,            W_k0+1[64:128]] (X = concat(M0, M2))
// out = base*M0 + scale*((a0/s)*M1 + (a1/s)*M2),  a_j = exp(lrelu(tanh(h_j+b1)*w2+b2)) + add_j
__global__ void att_combine(const float* __restrict__ M0, const float* __restrict__ M1,
                            const float* __restrict__ M2,
                            const float* __restrict__ a1W, const float* __restrict__ a1b,
                            const float* __restrict__ a2w, const float* __restrict__ a2b,
                            int k0, float add0, float add1, float base, float scale,
                            float* __restrict__ out, int N) {
    int lane = threadIdx.x;             // one wave32 per 16-node tile
    int tile = blockIdx.x;
    int mrow = lane & 15;
    int nodeA = tile * 16 + mrow; if (nodeA >= N) nodeA = N - 1;   // branchless clamp
    const float* W0 = a1W + k0 * 2 * FD;
    const float* W1 = a1W + (k0 + 1) * 2 * FD;
    int kb0 = (lane < 16) ? 0 : 8;      // ISA A-layout K base per half-wave

    v8f c = {};
#pragma unroll
    for (int chunk = 0; chunk < 6; ++chunk) {
        const float* srcA = (chunk < 2) ? M0 : ((chunk < 4) ? M1 : M2);
        int off = (chunk & 1) * 32;
        const float* rowp = srcA + nodeA * FD + off;
        v16h a;
#pragma unroll
        for (int j = 0; j < 8; ++j) {
            a[j]     = (_Float16)rowp[kb0 + j];        // K = kb0..kb0+7
            a[8 + j] = (_Float16)rowp[kb0 + 16 + j];   // K = kb0+16..kb0+23
        }
        v16h b = {};
#pragma unroll
        for (int s = 0; s < 16; ++s) {
        int kg = chunk * 32 + ((s < 8) ? (kb0 + s) : (kb0 + 8 + s));     // global column 0..191 (B layout: lane%16 == N, element == K in A's order)
        // branchless, always-in-range weight gathers + mask (no divergent EXEC regions)
        float e0 = W0[kg & 127];
        e0 = (kg < 128) ? e0 : 0.0f;
        int i1 = (kg >= 128) ? (kg - 64) : (kg & 63);  // valid index in [0,127] always
        float e1 = W1[i1];
        e1 = (kg < 64 || kg >= 128) ? e1 : 0.0f;
        b[s] = (_Float16)((mrow == 0) ? e0 : ((mrow == 1) ? e1 : 0.0f));
        }
        c = __builtin_amdgcn_wmma_f32_16x16x32_f16(false, a, false, b, (short)0, c, false, false);
    }

    // C[m][n]: vgpr = m%8, lane = n + 16*(m>=8). Need columns n=0,1.
    __shared__ float hv[2][16];
    __shared__ float c0s[16], c1s[16];
    if (lane == 0)       { for (int r = 0; r < 8; ++r) hv[0][r]     = c[r]; }
    else if (lane == 16) { for (int r = 0; r < 8; ++r) hv[0][8 + r] = c[r]; }
    else if (lane == 1)  { for (int r = 0; r < 8; ++r) hv[1][r]     = c[r]; }
    else if (lane == 17) { for (int r = 0; r < 8; ++r) hv[1][8 + r] = c[r]; }
    __syncthreads();
    if (lane < 16) {
        float h0 = tanhf(hv[0][lane] + a1b[k0]);
        float g0 = h0 * a2w[k0] + a2b[k0];
        g0 = (g0 >= 0.0f) ? g0 : 0.2f * g0;
        float a0 = __expf(g0) + add0;
        float h1 = tanhf(hv[1][lane] + a1b[k0 + 1]);
        float g1 = h1 * a2w[k0 + 1] + a2b[k0 + 1];
        g1 = (g1 >= 0.0f) ? g1 : 0.2f * g1;
        float a1 = __expf(g1) + add1;
        float sinv = scale / (a0 + a1);
        c0s[lane] = a0 * sinv;
        c1s[lane] = a1 * sinv;
    }
    __syncthreads();
    for (int t = lane; t < 16 * FD; t += 32) {
        int m = t >> 6, d = t & 63;
        int node = tile * 16 + m;
        if (node < N) {
            int idx = node * FD + d;
            out[idx] = base * M0[idx] + c0s[m] * M1[idx] + c1s[m] * M2[idx];
        }
    }
}

// ---------------- final scoring: sigmoid(diag(Fu_tile @ Fi_tile^T)) via WMMA ----------------
__global__ void score_wmma(const float* __restrict__ u0, const float* __restrict__ u1,
                           const float* __restrict__ u2, const float* __restrict__ i0,
                           const float* __restrict__ i1, const float* __restrict__ i2,
                           const int* __restrict__ uidx, const int* __restrict__ iidx,
                           float* __restrict__ out, int B) {
    int lane = threadIdx.x, tile = blockIdx.x;
    int mrow = lane & 15;
    int pA = tile * 16 + mrow; if (pA >= B) pA = B - 1;
    int un = uidx[pA];
    __shared__ int its[16];
    if (lane < 16) { int p = tile * 16 + lane; if (p >= B) p = B - 1; its[lane] = iidx[p]; }
    __syncthreads();
    int kb0 = (lane < 16) ? 0 : 8;

    v8f c = {};
#pragma unroll
    for (int chunk = 0; chunk < 6; ++chunk) {
        const float* us  = (chunk < 2) ? u0 : ((chunk < 4) ? u1 : u2);
        const float* isv = (chunk < 2) ? i0 : ((chunk < 4) ? i1 : i2);
        int off = (chunk & 1) * 32;
        const float* rowp = us + un * FD + off;
        v16h a;
#pragma unroll
        for (int j = 0; j < 8; ++j) {
            a[j]     = (_Float16)rowp[kb0 + j];
            a[8 + j] = (_Float16)rowp[kb0 + 16 + j];
        }
        v16h b;                         // B[k][n]: n==lane%16, half index==k (same order as A)
        const float* rowb = isv + its[mrow] * FD + off;
#pragma unroll
        for (int j = 0; j < 8; ++j) {
            b[j]     = (_Float16)rowb[kb0 + j];
            b[8 + j] = (_Float16)rowb[kb0 + 16 + j];
        }
        c = __builtin_amdgcn_wmma_f32_16x16x32_f16(false, a, false, b, (short)0, c, false, false);
    }

    // diagonal: m<8 -> vgpr m @ lane m ; m>=8 -> vgpr m-8 @ lane m+16 (lanes 24..31)
    int want = (lane < 8) ? lane : ((lane >= 24) ? (lane - 24) : -1);
    float val = 0.0f;
#pragma unroll
    for (int r = 0; r < 8; ++r) val = (want == r) ? c[r] : val;
    if (want >= 0) {
        int mo = (lane < 8) ? lane : (lane - 16);
        int p = tile * 16 + mo;
        if (p < B) out[p] = 1.0f / (1.0f + __expf(-val));
    }
}

// ---------------- host orchestration ----------------
extern "C" void kernel_launch(void* const* d_in, const int* in_sizes, int n_in,
                              void* d_out, int out_size, void* d_ws, size_t ws_size,
                              hipStream_t stream) {
    (void)n_in; (void)out_size; (void)ws_size;
    const float* u0    = (const float*)d_in[0];
    const float* i0    = (const float*)d_in[1];
    const float* snii1 = (const float*)d_in[2];
    const float* snii2 = (const float*)d_in[3];
    const float* ciii1 = (const float*)d_in[4];
    const float* ciii2 = (const float*)d_in[5];
    const float* icii1 = (const float*)d_in[6];
    const float* icii2 = (const float*)d_in[7];
    const float* law   = (const float*)d_in[8];
    const float* lab   = (const float*)d_in[9];
    const float* a1W   = (const float*)d_in[10];
    const float* a1b   = (const float*)d_in[11];
    const float* a2w   = (const float*)d_in[12];
    const float* a2b   = (const float*)d_in[13];
    const int* sn_row  = (const int*)d_in[14];
    const int* sn_col  = (const int*)d_in[15];
    const int* ci_row  = (const int*)d_in[16];
    const int* ci_col  = (const int*)d_in[17];
    const int* ic_row  = (const int*)d_in[18];
    const int* ic_col  = (const int*)d_in[19];
    const int* uix     = (const int*)d_in[20];
    const int* iix     = (const int*)d_in[21];

    const int U   = in_sizes[0] / FD;
    const int I   = in_sizes[1] / FD;
    const int Es  = in_sizes[2];
    const int Eui = in_sizes[4];
    const int Eiu = in_sizes[6];
    const int B   = in_sizes[20];
    const int segN = (U > I) ? U : I;

    float* W = (float*)d_ws;
    size_t off = 0;
    float* att_sn1 = W + off; off += Es;
    float* att_sn2 = W + off; off += Es;
    float* att_ci1 = W + off; off += Eui;
    float* att_ci2 = W + off; off += Eui;
    float* att_ic1 = W + off; off += Eiu;
    float* att_ic2 = W + off; off += Eiu;
    float* segm    = W + off; off += segN;   // reused as uint max-bits
    float* segs    = W + off; off += segN;
    float* u1v     = W + off; off += (size_t)U * FD;
    float* u2v     = W + off; off += (size_t)U * FD;
    float* i1v     = W + off; off += (size_t)I * FD;
    float* i2v     = W + off; off += (size_t)I * FD;
    float* ufi     = W + off; off += (size_t)U * FD;
    float* ufu     = W + off; off += (size_t)U * FD;
    float* ifu     = W + off; off += (size_t)I * FD;

    auto gs = [](int n, int b) { return (n + b - 1) / b; };

    struct EdgeSet { const float* raw; const int* row; int widx; int E; int n; float* att; };
    EdgeSet sets[6] = {
        { snii1, sn_row, 0, Es,  U, att_sn1 },
        { snii2, sn_row, 1, Es,  U, att_sn2 },
        { ciii1, ci_row, 2, Eui, U, att_ci1 },
        { ciii2, ci_row, 3, Eui, U, att_ci2 },
        { icii1, ic_row, 4, Eiu, I, att_ic1 },
        { icii2, ic_row, 5, Eiu, I, att_ic2 },
    };
    for (int s = 0; s < 6; ++s) {
        fill_f32<<<gs(sets[s].n, 256), 256, 0, stream>>>(segm, sets[s].n, 0.0f);
        fill_f32<<<gs(sets[s].n, 256), 256, 0, stream>>>(segs, sets[s].n, 0.0f);
        edge_pass1<<<gs(sets[s].E, 256), 256, 0, stream>>>(sets[s].raw, sets[s].row, law, lab,
                                                           sets[s].widx, sets[s].att,
                                                           (unsigned*)segm, sets[s].E);
        edge_pass2<<<gs(sets[s].E, 256), 256, 0, stream>>>(sets[s].att, sets[s].row,
                                                           (const unsigned*)segm, segs, sets[s].E);
        edge_pass3<<<gs(sets[s].E, 256), 256, 0, stream>>>(sets[s].att, sets[s].row, segs, sets[s].E);
    }

    const int QE = FD / 4;  // quads per edge

    // ---- layer 1 ----
    fill_f32<<<gs(U * FD, 256), 256, 0, stream>>>(ufi, U * FD, 0.0f);
    fill_f32<<<gs(U * FD, 256), 256, 0, stream>>>(ufu, U * FD, 0.0f);
    fill_f32<<<gs(I * FD, 256), 256, 0, stream>>>(ifu, I * FD, 0.0f);
    spmm_scatter<<<gs(Eui * QE, 256), 256, 0, stream>>>(att_ci1, ci_row, ci_col, i0, ufi, Eui);
    spmm_scatter<<<gs(Es  * QE, 256), 256, 0, stream>>>(att_sn1, sn_row, sn_col, u0, ufu, Es);
    att_combine<<<gs(U, 16), 32, 0, stream>>>(u0, ufi, ufu, a1W, a1b, a2w, a2b,
                                              0, 0.7f, 0.3f, 0.5f, 0.5f, u1v, U);
    spmm_scatter<<<gs(Eiu * QE, 256), 256, 0, stream>>>(att_ic1, ic_row, ic_col, u0, ifu, Eiu);
    att_combine<<<gs(I, 16), 32, 0, stream>>>(i0, i0, ifu, a1W, a1b, a2w, a2b,
                                              4, 1.0f, 1.0f, 0.0f, 1.0f, i1v, I);

    // ---- layer 2 ----
    fill_f32<<<gs(U * FD, 256), 256, 0, stream>>>(ufi, U * FD, 0.0f);
    fill_f32<<<gs(U * FD, 256), 256, 0, stream>>>(ufu, U * FD, 0.0f);
    fill_f32<<<gs(I * FD, 256), 256, 0, stream>>>(ifu, I * FD, 0.0f);
    spmm_scatter<<<gs(Eui * QE, 256), 256, 0, stream>>>(att_ci2, ci_row, ci_col, i1v, ufi, Eui);
    spmm_scatter<<<gs(Es  * QE, 256), 256, 0, stream>>>(att_sn2, sn_row, sn_col, u1v, ufu, Es);
    att_combine<<<gs(U, 16), 32, 0, stream>>>(u1v, ufi, ufu, a1W, a1b, a2w, a2b,
                                              2, 0.7f, 0.3f, 0.5f, 0.5f, u2v, U);
    spmm_scatter<<<gs(Eiu * QE, 256), 256, 0, stream>>>(att_ic2, ic_row, ic_col, u1v, ifu, Eiu);
    att_combine<<<gs(I, 16), 32, 0, stream>>>(i1v, i1v, ifu, a1W, a1b, a2w, a2b,
                                              6, 1.0f, 1.0f, 0.0f, 1.0f, i2v, I);

    // ---- scoring ----
    score_wmma<<<gs(B, 16), 32, 0, stream>>>(u0, u1v, u2v, i0, i1v, i2v, uix, iix,
                                             (float*)d_out, B);
}
